// SkipTransformer_11888469475435
// MI455X (gfx1250) — matrix operand order repaired and checked by reading, not compiled, not run on hardware
//
#include <hip/hip_runtime.h>

// ---------------------------------------------------------------------------
// SkipTransformer (Point Transformer block) for gfx1250 / MI455X.
// Heavy math (attn MLP: 137 GFLOP) fused with bf16 WMMA, f32 accumulate.
// Round 2: 4 points (64 columns) per attn workgroup -> 4x weight reuse per
// A-tile load, quartering L2->WGP weight traffic (8.4 GB -> 2.1 GB).
// Peak LDS ~228 KB (legal: CDNA5 allows 320 KB per workgroup).
// ---------------------------------------------------------------------------

#define BN_EPS 1e-5f
constexpr int Bn = 2, Nn = 4096, Cc = 128, Dm = 256, Kk = 16, Ph = 64, Ah = 1024;
constexpr int NP = 4;                    // points per attn workgroup
constexpr int NCOL = NP * Kk;            // 64 columns

typedef __attribute__((ext_vector_type(16))) __bf16 v16bf;
typedef __attribute__((ext_vector_type(8)))  __bf16 v8bf;
typedef __attribute__((ext_vector_type(8)))  float  v8f;

__device__ __forceinline__ v16bf pack16(const __bf16* p0, const __bf16* p1) {
  v8bf a = *(const v8bf*)p0;
  v8bf b = *(const v8bf*)p1;
  v16bf r;
#pragma unroll
  for (int i = 0; i < 8; ++i) { r[i] = a[i]; r[8 + i] = b[i]; }
  return r;
}

// A tile (16 x 32) from row-major bf16 weights W[Mtot x Ktot] at (m0, k0).
// ISA 16-bit A layout: lanes 0-15 -> M, K halves {0..7|16..23}; lanes 16-31 +8.
__device__ __forceinline__ v16bf loadA(const __bf16* W, int Ktot, int m0, int k0, int lane) {
  const int M  = m0 + (lane & 15);
  const int hb = (lane >> 4) << 3;           // 0 or 8
  const __bf16* p = W + M * Ktot + k0 + hb;
  return pack16(p, p + 16);
}

// B tile (32 x 16) from column-major LDS X[col][Ktot].
// ISA B layout: lanes 0-15 hold K {0..7|16..23} (col = lane), lanes 16-31 +8.
__device__ __forceinline__ v16bf loadB(const __bf16* X, int Ktot, int k0, int lane) {
  const int col = lane & 15;
  const int kb  = k0 + ((lane >> 4) << 3);   // +8 for upper half-wave
  const __bf16* p = X + col * Ktot + kb;
  return pack16(p, p + 16);
}

__device__ __forceinline__ v8f wmma_bf16(v16bf a, v16bf b, v8f c) {
  return __builtin_amdgcn_wmma_f32_16x16x32_bf16(false, a, false, b, (short)0, c, false, false);
}

// ---------------------------------------------------------------------------
// prep: bf16 weight conversion + BN folding
// ---------------------------------------------------------------------------
__global__ void prep_kernel(const float* w_v1, const float* w_v2, const float* w_vs,
                            const float* w_k,  const float* w_q,  const float* w_val,
                            const float* w_a1, const float* b_a1, const float* g_a,
                            const float* be_a, const float* m_a,  const float* v_a,
                            const float* w_a2,
                            const float* b_p1, const float* g_p, const float* be_p,
                            const float* m_p,  const float* v_p,
                            __bf16* Wv1, __bf16* Wv2, __bf16* Wvs,
                            __bf16* Wk,  __bf16* Wq,  __bf16* Wval,
                            __bf16* Wa1, __bf16* Wa2, float* ba1f,
                            float* sp, float* bpf) {
  int i = blockIdx.x * 256 + threadIdx.x;
  if (i < 32768) { Wv1[i] = (__bf16)w_v1[i]; return; } i -= 32768;
  if (i < 16384) { Wv2[i] = (__bf16)w_v2[i]; return; } i -= 16384;
  if (i < 32768) { Wvs[i] = (__bf16)w_vs[i]; return; } i -= 32768;
  if (i < 32768) { Wk[i]  = (__bf16)w_k[i];  return; } i -= 32768;
  if (i < 32768) { Wq[i]  = (__bf16)w_q[i];  return; } i -= 32768;
  if (i < 32768) { Wval[i]= (__bf16)w_val[i];return; } i -= 32768;
  if (i < 262144){ Wa2[i] = (__bf16)w_a2[i]; return; } i -= 262144;
  if (i < 262144){ int o = i >> 8;
                   float s = g_a[o] * rsqrtf(v_a[o] + BN_EPS);
                   Wa1[i] = (__bf16)(s * w_a1[i]); return; } i -= 262144;
  if (i < 1024)  { float s = g_a[i] * rsqrtf(v_a[i] + BN_EPS);
                   ba1f[i] = s * (b_a1[i] - m_a[i]) + be_a[i]; return; } i -= 1024;
  if (i < 64)    { float s = g_p[i] * rsqrtf(v_p[i] + BN_EPS);
                   sp[i] = s; bpf[i] = s * (b_p1[i] - m_p[i]) + be_p[i]; }
}

// ---------------------------------------------------------------------------
// knn: top-16 nearest (squared L2, includes self), LDS-tiled candidates
// ---------------------------------------------------------------------------
__global__ void knn_kernel(const float* pos, int* idx) {
  const int t = threadIdx.x;
  const int gid = blockIdx.x * 256 + t;
  const int b = gid >> 12, n = gid & (Nn - 1);
  const float* pb = pos + b * 3 * Nn;
  const float qx = pb[n], qy = pb[Nn + n], qz = pb[2 * Nn + n];
  float bd[Kk]; int bi[Kk];
#pragma unroll
  for (int j = 0; j < Kk; ++j) { bd[j] = 3.0e38f; bi[j] = 0; }
  __shared__ float cx[256], cy[256], cz[256];
  for (int tile = 0; tile < Nn; tile += 256) {
    __syncthreads();
    cx[t] = pb[tile + t]; cy[t] = pb[Nn + tile + t]; cz[t] = pb[2 * Nn + tile + t];
    __syncthreads();
    for (int j = 0; j < 256; ++j) {
      float dx = qx - cx[j], dy = qy - cy[j], dz = qz - cz[j];
      float d = dx * dx + dy * dy + dz * dz;
      if (d < bd[Kk - 1]) {
        float dd = d; int ii = tile + j;
#pragma unroll
        for (int r = 0; r < Kk; ++r) {
          if (dd < bd[r]) { float td = bd[r]; int ti = bi[r];
                            bd[r] = dd; bi[r] = ii; dd = td; ii = ti; }
        }
      }
    }
  }
  int* op = idx + (b * Nn + n) * Kk;
#pragma unroll
  for (int j = 0; j < Kk; ++j) op[j] = bi[j];
}

// ---------------------------------------------------------------------------
// value MLP (MlpRes): V = Wv2*relu(Wv1*[key;query]+b1) + Wvs*[key;query] + b2+bs
// 16 points per workgroup, 8 waves each own one 16-row M tile. Point-major out.
// ---------------------------------------------------------------------------
__global__ void value_kernel(const float* key, const float* query,
                             const __bf16* Wv1, const __bf16* Wv2, const __bf16* Wvs,
                             const float* b_v1, const float* b_v2, const float* b_vs,
                             float* value_pm) {
  const int t = threadIdx.x, lane = t & 31, wave = t >> 5;
  const int b = blockIdx.x / (Nn / 16);
  const int n0 = (blockIdx.x % (Nn / 16)) * 16;
  __shared__ __bf16 xv[16 * 256];
  __shared__ __bf16 h1[16 * 128];
  { // load X columns (concat key,query), column-major bf16
    const int col = t & 15, c0 = (t >> 4) * 16, n = n0 + col;
#pragma unroll
    for (int j = 0; j < 16; ++j) {
      int c = c0 + j;
      float x = (c < 128) ? key[(b * 128 + c) * Nn + n]
                          : query[(b * 128 + (c - 128)) * Nn + n];
      xv[col * 256 + c] = (__bf16)x;
    }
  }
  __syncthreads();
  const int m0 = wave * 16, col = lane & 15, hl = lane >> 4;
  v8f acc = {};
  for (int k0 = 0; k0 < 256; k0 += 32)
    acc = wmma_bf16(loadA(Wv1, 256, m0, k0, lane), loadB(xv, 256, k0, lane), acc);
#pragma unroll
  for (int r = 0; r < 8; ++r) {
    int m = m0 + r + 8 * hl;
    h1[col * 128 + m] = (__bf16)fmaxf(acc[r] + b_v1[m], 0.f);
  }
  __syncthreads();
  v8f acc2 = {};
  for (int k0 = 0; k0 < 128; k0 += 32)
    acc2 = wmma_bf16(loadA(Wv2, 128, m0, k0, lane), loadB(h1, 128, k0, lane), acc2);
  for (int k0 = 0; k0 < 256; k0 += 32)
    acc2 = wmma_bf16(loadA(Wvs, 256, m0, k0, lane), loadB(xv, 256, k0, lane), acc2);
#pragma unroll
  for (int r = 0; r < 8; ++r) {
    int m = m0 + r + 8 * hl;
    value_pm[(b * Nn + n0 + col) * 128 + m] = acc2[r] + b_v2[m] + b_vs[m];
  }
}

// ---------------------------------------------------------------------------
// proj: out_pm(B,N,256) = W(256x128) * in_col + bias. in layout flag.
// ---------------------------------------------------------------------------
__global__ void proj_kernel(const float* in, int point_major,
                            const __bf16* W, const float* bias, float* out_pm) {
  const int t = threadIdx.x, lane = t & 31, wave = t >> 5;
  const int b = blockIdx.x / (Nn / 16);
  const int n0 = (blockIdx.x % (Nn / 16)) * 16;
  __shared__ __bf16 xin[16 * 128];
  { const int col = t & 15, c0 = (t >> 4) * 8, n = n0 + col;
#pragma unroll
    for (int j = 0; j < 8; ++j) {
      int c = c0 + j;
      float x = point_major ? in[(b * Nn + n) * 128 + c]
                            : in[(b * 128 + c) * Nn + n];
      xin[col * 128 + c] = (__bf16)x;
    }
  }
  __syncthreads();
  const int col = lane & 15, hl = lane >> 4;
#pragma unroll
  for (int i = 0; i < 2; ++i) {
    const int m0 = (wave * 2 + i) * 16;
    v8f acc = {};
    for (int k0 = 0; k0 < 128; k0 += 32)
      acc = wmma_bf16(loadA(W, 128, m0, k0, lane), loadB(xin, 128, k0, lane), acc);
#pragma unroll
    for (int r = 0; r < 8; ++r) {
      int m = m0 + r + 8 * hl;
      out_pm[(b * Nn + n0 + col) * 256 + m] = acc[r] + bias[m];
    }
  }
}

// ---------------------------------------------------------------------------
// attn: one workgroup (8 waves) per 4 points. Fused attention MLP with WMMA:
// each A (weight) tile load feeds 4 WMMAs (one per 16-col point tile) -> 4x
// weight reuse. Softmax over K, aggregation v4 = vf + pe, fused conv_end +
// identity. LDS regions phase-aliased; peak ~228 KB (<320 KB CDNA5 limit).
// ---------------------------------------------------------------------------
__global__ __launch_bounds__(256)
void attn_kernel(const float* pos, const int* idx,
                 const float* kf, const float* qf, const float* vf,
                 const float* w_p1, const float* sp, const float* bpf,
                 const float* w_p2, const float* b_p2,
                 const __bf16* Wa1, const float* ba1f,
                 const __bf16* Wa2, const float* b_a2,
                 const float* w_e, const float* b_e,
                 const float* value_pm, float* out) {
  const int t = threadIdx.x, lane = t & 31, wave = t >> 5;
  const int b = blockIdx.x / (Nn / NP);
  const int n0 = (blockIdx.x % (Nn / NP)) * NP;

  // LDS layout (phase-aliased):
  //  hS   [ 0      .. 131072) bf16 64x1024  ph4(w) ph5(r)
  //  peS  [131072  .. 163840) bf16 64x256   ph3(w) ph6(r)
  //  xkS  [163840  .. 196608) bf16 64x256   ph3(w) ph4(r)
  //  ysS  [163840  .. 229376) f32 256x64    ph5(w) ph6(r)  (aliases xk+ph)
  //  phS  [196608  .. 212992) f32 64x64     ph2(w) ph3(r)
  //  aggS [229376  .. 233472) f32 4x256     ph6(w) ph7(r)
  //  idxS [233472  .. 233728) 64 ints
  __shared__ __attribute__((aligned(16))) char smem[233728];
  __bf16* hS   = (__bf16*)(smem + 0);
  __bf16* peS  = (__bf16*)(smem + 131072);
  __bf16* xkS  = (__bf16*)(smem + 163840);
  float*  ysS  = (float*) (smem + 163840);
  float*  phS  = (float*) (smem + 196608);
  float*  aggS = (float*) (smem + 229376);
  int*    idxS = (int*)   (smem + 233472);

  if (t < NCOL) idxS[t] = idx[(b * Nn + n0 + (t >> 4)) * Kk + (t & 15)];
  __syncthreads();

  { // phase 2: pos-MLP hidden (64) per column, BN folded, relu
    const int c = t >> 2, j = t & 3;       // 4 threads per column
    const int n = n0 + (c >> 4);
    const int ik = idxS[c];
    const float* pb = pos + b * 3 * Nn;
    const float prx = pb[n] - pb[ik];
    const float pry = pb[Nn + n] - pb[Nn + ik];
    const float prz = pb[2 * Nn + n] - pb[2 * Nn + ik];
#pragma unroll
    for (int jj = 0; jj < 16; ++jj) {
      int o = j * 16 + jj;
      float dot = w_p1[o * 3] * prx + w_p1[o * 3 + 1] * pry + w_p1[o * 3 + 2] * prz;
      phS[c * 64 + o] = fmaxf(sp[o] * dot + bpf[o], 0.f);
    }
  }
  __syncthreads();

  { // phase 3: pe (pos-MLP out) + X = qk_rel + pe (bf16, column-major)
    const int c = t >> 2, j = t & 3;       // 4 threads per column, 64 m each
    const float* qc = qf + (b * Nn + n0 + (c >> 4)) * 256;
    const float* kc = kf + (b * Nn + idxS[c]) * 256;
    const float* ph = phS + c * 64;
#pragma unroll 4
    for (int mm = 0; mm < 64; ++mm) {
      int m = j * 64 + mm;
      float pe = b_p2[m];
      const float* wr = w_p2 + m * 64;
#pragma unroll
      for (int jj = 0; jj < 64; ++jj) pe += wr[jj] * ph[jj];
      peS[c * 256 + m] = (__bf16)pe;
      xkS[c * 256 + m] = (__bf16)(qc[m] - kc[m] + pe);
    }
  }
  __syncthreads();

  const int col = lane & 15, hl = lane >> 4;
  __builtin_prefetch(Wa2 + wave * 32 * 1024, 0, 1);  // warm layer-2 weights

  // phase 4: H(1024x64) = relu(Wa1f * X + ba1f); 4 B-tiles per A-tile load
#pragma unroll
  for (int mt = 0; mt < 8; ++mt) {
    const int m0 = (wave + mt * 8) * 16;
    v8f acc[NP] = {{}, {}, {}, {}};
    for (int k0 = 0; k0 < 256; k0 += 32) {
      v16bf a = loadA(Wa1, 256, m0, k0, lane);
#pragma unroll
      for (int nt = 0; nt < NP; ++nt)
        acc[nt] = wmma_bf16(a, loadB(xkS + nt * 16 * 256, 256, k0, lane), acc[nt]);
    }
    float bias[8];
#pragma unroll
    for (int r = 0; r < 8; ++r) bias[r] = ba1f[m0 + r + 8 * hl];
#pragma unroll
    for (int nt = 0; nt < NP; ++nt)
#pragma unroll
      for (int r = 0; r < 8; ++r) {
        int m = m0 + r + 8 * hl;
        hS[(nt * 16 + col) * 1024 + m] = (__bf16)fmaxf(acc[nt][r] + bias[r], 0.f);
      }
  }
  __syncthreads();

  // phase 5: Y(256x64) = Wa2 * H + b_a2 (attention logits)
#pragma unroll
  for (int mt = 0; mt < 2; ++mt) {
    const int m0 = (wave * 2 + mt) * 16;
    v8f acc[NP] = {{}, {}, {}, {}};
    for (int k0 = 0; k0 < 1024; k0 += 32) {
      v16bf a = loadA(Wa2, 1024, m0, k0, lane);
#pragma unroll
      for (int nt = 0; nt < NP; ++nt)
        acc[nt] = wmma_bf16(a, loadB(hS + nt * 16 * 1024, 1024, k0, lane), acc[nt]);
    }
#pragma unroll
    for (int nt = 0; nt < NP; ++nt)
#pragma unroll
      for (int r = 0; r < 8; ++r) {
        int m = m0 + r + 8 * hl;
        ysS[m * NCOL + nt * 16 + col] = acc[nt][r] + b_a2[m];
      }
  }
  __syncthreads();

  { // phase 6: softmax over K + aggregate v4 = vf + pe, per point
    const int m = t;
#pragma unroll
    for (int p = 0; p < NP; ++p) {
      const float* yr = ysS + m * NCOL + p * 16;
      float mx = yr[0];
#pragma unroll
      for (int k = 1; k < Kk; ++k) mx = fmaxf(mx, yr[k]);
      float w[Kk], s = 0.f;
#pragma unroll
      for (int k = 0; k < Kk; ++k) { w[k] = __expf(yr[k] - mx); s += w[k]; }
      const float inv = 1.f / s;
      const float vfm = vf[(b * Nn + n0 + p) * 256 + m];
      float agg = 0.f;
#pragma unroll
      for (int k = 0; k < Kk; ++k)
        agg += (w[k] * inv) * (vfm + (float)peS[(p * 16 + k) * 256 + m]);
      aggS[p * 256 + m] = agg;
    }
  }
  __syncthreads();

  if (t < Cc) { // phase 7: conv_end + identity, per point
    const float* wr = w_e + t * 256;
#pragma unroll
    for (int p = 0; p < NP; ++p) {
      float r = b_e[t];
      const float* ag = aggS + p * 256;
#pragma unroll 8
      for (int c = 0; c < 256; ++c) r += wr[c] * ag[c];
      r += value_pm[(b * Nn + n0 + p) * 128 + t];
      out[(b * Cc + t) * Nn + n0 + p] = r;
    }
  }
}

// ---------------------------------------------------------------------------
extern "C" void kernel_launch(void* const* d_in, const int* in_sizes, int n_in,
                              void* d_out, int out_size, void* d_ws, size_t ws_size,
                              hipStream_t stream) {
  const float* pos   = (const float*)d_in[0];
  const float* key   = (const float*)d_in[1];
  const float* query = (const float*)d_in[2];
  const float* w_v1  = (const float*)d_in[3];  const float* b_v1 = (const float*)d_in[4];
  const float* w_v2  = (const float*)d_in[5];  const float* b_v2 = (const float*)d_in[6];
  const float* w_vs  = (const float*)d_in[7];  const float* b_vs = (const float*)d_in[8];
  const float* w_k   = (const float*)d_in[9];  const float* b_k  = (const float*)d_in[10];
  const float* w_q   = (const float*)d_in[11]; const float* b_q  = (const float*)d_in[12];
  const float* w_val = (const float*)d_in[13]; const float* b_val= (const float*)d_in[14];
  const float* w_p1  = (const float*)d_in[15]; const float* b_p1 = (const float*)d_in[16];
  const float* g_p   = (const float*)d_in[17]; const float* be_p = (const float*)d_in[18];
  const float* m_p   = (const float*)d_in[19]; const float* v_p  = (const float*)d_in[20];
  const float* w_p2  = (const float*)d_in[21]; const float* b_p2 = (const float*)d_in[22];
  const float* w_a1  = (const float*)d_in[23]; const float* b_a1 = (const float*)d_in[24];
  const float* g_a   = (const float*)d_in[25]; const float* be_a = (const float*)d_in[26];
  const float* m_a   = (const float*)d_in[27]; const float* v_a  = (const float*)d_in[28];
  const float* w_a2  = (const float*)d_in[29]; const float* b_a2 = (const float*)d_in[30];
  const float* w_e   = (const float*)d_in[31]; const float* b_e  = (const float*)d_in[32];
  (void)in_sizes; (void)n_in; (void)out_size; (void)ws_size;

  // workspace layout (~31 MB), 256B-aligned segments
  char* ws = (char*)d_ws;
  size_t off = 0;
  auto take = [&](size_t bytes) { char* p = ws + off; off = (off + bytes + 255) & ~(size_t)255; return p; };
  int*    idx      = (int*)   take((size_t)Bn * Nn * Kk * 4);
  float*  value_pm = (float*) take((size_t)Bn * Nn * Cc * 4);
  float*  kf_pm    = (float*) take((size_t)Bn * Nn * Dm * 4);
  float*  qf_pm    = (float*) take((size_t)Bn * Nn * Dm * 4);
  float*  vf_pm    = (float*) take((size_t)Bn * Nn * Dm * 4);
  __bf16* Wv1      = (__bf16*)take(128 * 256 * 2);
  __bf16* Wv2      = (__bf16*)take(128 * 128 * 2);
  __bf16* Wvs      = (__bf16*)take(128 * 256 * 2);
  __bf16* Wk       = (__bf16*)take(256 * 128 * 2);
  __bf16* Wq       = (__bf16*)take(256 * 128 * 2);
  __bf16* Wval     = (__bf16*)take(256 * 128 * 2);
  __bf16* Wa1      = (__bf16*)take(1024 * 256 * 2);
  __bf16* Wa2      = (__bf16*)take(256 * 1024 * 2);
  float*  ba1f     = (float*) take(1024 * 4);
  float*  sp       = (float*) take(64 * 4);
  float*  bpf      = (float*) take(64 * 4);

  const int prep_elems = 32768 * 5 + 16384 + 262144 * 2 + 1024 + 64;
  prep_kernel<<<(prep_elems + 255) / 256, 256, 0, stream>>>(
      w_v1, w_v2, w_vs, w_k, w_q, w_val, w_a1, b_a1, g_a, be_a, m_a, v_a, w_a2,
      b_p1, g_p, be_p, m_p, v_p,
      Wv1, Wv2, Wvs, Wk, Wq, Wval, Wa1, Wa2, ba1f, sp, bpf);

  knn_kernel<<<(Bn * Nn) / 256, 256, 0, stream>>>(pos, idx);

  value_kernel<<<(Bn * Nn) / 16, 256, 0, stream>>>(
      key, query, Wv1, Wv2, Wvs, b_v1, b_v2, b_vs, value_pm);

  proj_kernel<<<(Bn * Nn) / 16, 256, 0, stream>>>(key,      0, Wk,   b_k,   kf_pm);
  proj_kernel<<<(Bn * Nn) / 16, 256, 0, stream>>>(query,    0, Wq,   b_q,   qf_pm);
  proj_kernel<<<(Bn * Nn) / 16, 256, 0, stream>>>(value_pm, 1, Wval, b_val, vf_pm);

  attn_kernel<<<(Bn * Nn) / NP, 256, 0, stream>>>(
      pos, idx, kf_pm, qf_pm, vf_pm,
      w_p1, sp, bpf, w_p2, b_p2,
      Wa1, ba1f, Wa2, b_a2, w_e, b_e, value_pm, (float*)d_out);
}
